// ISONET_34505767256121
// MI455X (gfx1250) — matrix operand order repaired and checked by reading, not compiled, not run on hardware
//
#include <hip/hip_runtime.h>
#include <hip/hip_bf16.h>

// ---------------------------------------------------------------------------
// Problem constants (match reference)
// ---------------------------------------------------------------------------
constexpr int G_GRAPHS = 2048;
constexpr int NPG = 64;       // nodes per graph
constexpr int EPG = 256;      // edges per graph
constexpr int MSGD = 128;
constexpr int MAXS_T = 304;   // MAX_SET=300 padded to 16
constexpr int S_LD = 320;     // S leading dim (K-tiles of 32 cover 0..319)

typedef __attribute__((ext_vector_type(16))) __bf16 v16bf;
typedef __attribute__((ext_vector_type(8)))  __bf16 v8bf;
typedef __attribute__((ext_vector_type(8)))  float  v8f;

// ---------------------------------------------------------------------------
// WMMA helpers (CDNA5 bf16 16x16x32, A/B k = 16*(lane/16) + i, contiguous)
// ---------------------------------------------------------------------------
__device__ __forceinline__ int lane_id() { return threadIdx.x & 31; }

__device__ __forceinline__ v8f wmma_bf16(v16bf a, v16bf b, v8f c) {
  return __builtin_amdgcn_wmma_f32_16x16x32_bf16(false, a, false, b, (short)0, c,
                                                 false, false);
}

// A fragment: 16(M) x 32(K) slice from row-major bf16 (ld multiple of 8,
// 16B-aligned rows). Two 16-byte vector loads per lane.
__device__ __forceinline__ v16bf a_frag(const __bf16* src, int ld) {
  const int l = lane_id();
  const __bf16* p = src + (l & 15) * ld + ((l >> 4) << 4);
  v8bf lo = *(const v8bf*)(p);
  v8bf hi = *(const v8bf*)(p + 8);
  return __builtin_shufflevector(lo, hi, 0, 1, 2, 3, 4, 5, 6, 7, 8, 9, 10, 11,
                                 12, 13, 14, 15);
}

// A fragment converted from f32 source, only elements 8..15 zeroed for k >= kmax (0..7 always loaded)
__device__ __forceinline__ v16bf a_frag_f32(const float* src, int ld, int kmax) {
  const int l = lane_id();
  const int h = (l >> 4) << 4;
  const float* p = src + (l & 15) * ld + h;
  v16bf a;
#pragma unroll
  for (int i = 0; i < 8; ++i) a[i] = (__bf16)p[i];
#pragma unroll
  for (int i = 0; i < 8; ++i)
    a[i + 8] = (8 + h + i < kmax) ? (__bf16)p[8 + i] : (__bf16)0.0f;
  return a;
}

// B fragment from fragment-packed storage: [tile][lane][16] bf16, one 32B load
__device__ __forceinline__ v16bf b_frag_p(const __bf16* W, int kt, int nt,
                                          int ntiles) {
  return *(const v16bf*)(W + (((kt * ntiles + nt) * 32 + lane_id()) << 4));
}

__device__ __forceinline__ v16bf frag_lds_p(const __bf16* base, int tile) {
  return *(const v16bf*)(base + ((tile * 32 + lane_id()) << 4));
}

// B fragment where the source is N-major (row-major N x K): B = src^T.
__device__ __forceinline__ v16bf bT_frag(const __bf16* src, int ld) {
  const int l = lane_id();
  const __bf16* p = src + (l & 15) * ld + ((l >> 4) << 4);
  v8bf lo = *(const v8bf*)(p);
  v8bf hi = *(const v8bf*)(p + 8);
  return __builtin_shufflevector(lo, hi, 0, 1, 2, 3, 4, 5, 6, 7, 8, 9, 10, 11,
                                 12, 13, 14, 15);
}

// C fragment initialized with a per-column bias (broadcast down rows)
__device__ __forceinline__ v8f bias_frag(const float* bias) {
  const float bv = bias[lane_id() & 15];
  v8f c;
#pragma unroll
  for (int r = 0; r < 8; ++r) c[r] = bv;
  return c;
}

// Store D tile (16x16 f32 accumulators) to bf16, row = r + 8*(lane>=16)
__device__ __forceinline__ void store_tile_bf16(__bf16* dst, int ld, v8f c, bool relu) {
  const int l = lane_id();
  const int m0 = (l >> 4) << 3;
  const int n = l & 15;
#pragma unroll
  for (int r = 0; r < 8; ++r) {
    float x = c[r];
    if (relu) x = fmaxf(x, 0.0f);
    dst[(m0 + r) * ld + n] = (__bf16)x;
  }
}

// ---------------------------------------------------------------------------
// Kernel A: one workgroup per graph, whole 5-round GNN fused in LDS
// ---------------------------------------------------------------------------
struct GnnParams {
  const float* node_f;
  const float* edge_f;
  const int* from_idx;
  const int* to_idx;
  const __bf16 *Wen, *Wee, *Wm1, *Wm2, *Wr1, *Wr2, *Wu1, *Wu2;  // frag-packed
  const float *ben, *bee, *bm1, *bm2, *br1, *br2, *bu1, *bu2;
  __bf16* edge_emb;  // [E x 128] bf16 output
};

// LDS layout (bytes)
constexpr int A_OFF_FROM = 0;        // int[256]
constexpr int A_OFF_TO   = 1024;     // int[256]
constexpr int A_OFF_NODE = 2048;     // bf16[64*64]
constexpr int A_OFF_EDGE = 10240;    // bf16[256*64]
constexpr int A_OFF_MSG  = 43008;    // float[64*128]
constexpr int A_OFF_AT   = 75776;    // bf16[8][16*192]  per-wave A scratch
constexpr int A_OFF_H    = 124928;   // bf16[8][2][16*128] per-wave hidden
constexpr int A_SMEM     = 190464;

// Build 16x192 A tile = [node_h[first] | node_h[second] | edge_h] (dword copies)
__device__ __forceinline__ void build_edge_A(__bf16* A, const __bf16* sNode,
                                             const __bf16* sEdge, const int* first,
                                             const int* second, int t) {
  const int l = lane_id();
  unsigned* Ad = (unsigned*)A;
  const unsigned* Nd = (const unsigned*)sNode;
  const unsigned* Ed = (const unsigned*)sEdge;
  for (int r = 0; r < 16; ++r) {
    const int e = t * 16 + r;
    Ad[r * 96 + l]      = Nd[first[e] * 32 + l];
    Ad[r * 96 + 32 + l] = Nd[second[e] * 32 + l];
    Ad[r * 96 + 64 + l] = Ed[e * 32 + l];
  }
}

// 16x192 @ 192x128 + bias, relu -> bf16 H (ld 128). A fragments hoisted.
__device__ __forceinline__ void mlp_layer1_192(__bf16* H, const __bf16* A,
                                               const __bf16* W, const float* bias) {
  v16bf af[6];
#pragma unroll
  for (int ks = 0; ks < 6; ++ks) af[ks] = a_frag(A + ks * 32, 192);
#pragma unroll
  for (int nt = 0; nt < 8; ++nt) {
    v8f c = bias_frag(bias + nt * 16);
#pragma unroll
    for (int ks = 0; ks < 6; ++ks)
      c = wmma_bf16(af[ks], b_frag_p(W, ks, nt, 8), c);
    store_tile_bf16(H + nt * 16, 128, c, true);
  }
}

__global__ void __launch_bounds__(256)
gnn_graph_kernel(GnnParams P) {
  extern __shared__ char smem[];
  int*    sFrom = (int*)(smem + A_OFF_FROM);
  int*    sTo   = (int*)(smem + A_OFF_TO);
  __bf16* sNode = (__bf16*)(smem + A_OFF_NODE);
  __bf16* sEdge = (__bf16*)(smem + A_OFF_EDGE);
  float*  sMsg  = (float*)(smem + A_OFF_MSG);
  __bf16* sAall = (__bf16*)(smem + A_OFF_AT);
  __bf16* sHall = (__bf16*)(smem + A_OFF_H);

  const int g = blockIdx.x;
  const int tid = threadIdx.x;
  const int wave = tid >> 5;
  __bf16* sA  = sAall + wave * (16 * 192);
  __bf16* sH0 = sHall + wave * (2 * 16 * 128);
  __bf16* sH1 = sH0 + 16 * 128;

  // local edge endpoints (graph-local node ids)
  sFrom[tid] = P.from_idx[g * EPG + tid] - g * NPG;
  sTo[tid]   = P.to_idx[g * EPG + tid] - g * NPG;

  // -------- encoders --------
  if (wave < 4) {  // node encoder: 64x32 @ 32x64
    const int mt = wave;
    v16bf a = a_frag_f32(P.node_f + (g * NPG + mt * 16) * 32, 32, 32);
#pragma unroll
    for (int nt = 0; nt < 4; ++nt) {
      v8f c = bias_frag(P.ben + nt * 16);
      c = wmma_bf16(a, b_frag_p(P.Wen, 0, nt, 4), c);
      store_tile_bf16(sNode + mt * 16 * 64 + nt * 16, 64, c, false);
    }
  }
  for (int t = wave; t < 16; t += 8) {  // edge encoder: 256x16 @ 16x64 (K pad 32)
    v16bf a = a_frag_f32(P.edge_f + (g * EPG + t * 16) * 16, 16, 16);
#pragma unroll
    for (int nt = 0; nt < 4; ++nt) {
      v8f c = bias_frag(P.bee + nt * 16);
      c = wmma_bf16(a, b_frag_p(P.Wee, 0, nt, 4), c);
      store_tile_bf16(sEdge + t * 16 * 64 + nt * 16, 64, c, false);
    }
  }
  __syncthreads();

  // -------- 5 propagation rounds, fully in LDS --------
  for (int round = 0; round < 5; ++round) {
    for (int i = tid; i < NPG * MSGD; i += 256) sMsg[i] = 0.0f;
    __syncthreads();

    for (int t = wave; t < 16; t += 8) {
      build_edge_A(sA, sNode, sEdge, sFrom, sTo, t);
      mlp_layer1_192(sH0, sA, P.Wm1, P.bm1);
      build_edge_A(sA, sNode, sEdge, sTo, sFrom, t);
      mlp_layer1_192(sH1, sA, P.Wr1, P.br1);

      v16bf h0[4], h1[4];
#pragma unroll
      for (int ks = 0; ks < 4; ++ks) {
        h0[ks] = a_frag(sH0 + ks * 32, 128);
        h1[ks] = a_frag(sH1 + ks * 32, 128);
      }
#pragma unroll
      for (int nt = 0; nt < 8; ++nt) {
        v8f c0 = bias_frag(P.bm2 + nt * 16);
        v8f c1 = bias_frag(P.br2 + nt * 16);
#pragma unroll
        for (int ks = 0; ks < 4; ++ks) {
          c0 = wmma_bf16(h0[ks], b_frag_p(P.Wm2, ks, nt, 8), c0);
          c1 = wmma_bf16(h1[ks], b_frag_p(P.Wr2, ks, nt, 8), c1);
        }
        // segment_sum via LDS float atomics (ds_add_f32)
        const int m0 = (lane_id() >> 4) << 3;
        const int n = lane_id() & 15;
#pragma unroll
        for (int r = 0; r < 8; ++r) {
          const int e = t * 16 + m0 + r;
          atomicAdd(&sMsg[sTo[e] * MSGD + nt * 16 + n], c0[r]);
          atomicAdd(&sMsg[sFrom[e] * MSGD + nt * 16 + n], c1[r]);
        }
      }
    }
    __syncthreads();

    // node update: [msgs | node_h] (64x192) -> 128 relu -> 64
    if (wave < 4) {
      const int mt = wave;
      const int l = lane_id();
      for (int r = 0; r < 16; ++r) {
        const int node = mt * 16 + r;
#pragma unroll
        for (int cc = 0; cc < 4; ++cc) {
          const int c = l + cc * 32;
          sA[r * 192 + c] = (__bf16)sMsg[node * MSGD + c];
        }
        ((unsigned*)sA)[r * 96 + 64 + l] = ((const unsigned*)sNode)[node * 32 + l];
      }
      mlp_layer1_192(sH0, sA, P.Wu1, P.bu1);
      v16bf h0[4];
#pragma unroll
      for (int ks = 0; ks < 4; ++ks) h0[ks] = a_frag(sH0 + ks * 32, 128);
#pragma unroll
      for (int nt = 0; nt < 4; ++nt) {
        v8f c = bias_frag(P.bu2 + nt * 16);
#pragma unroll
        for (int ks = 0; ks < 4; ++ks)
          c = wmma_bf16(h0[ks], b_frag_p(P.Wu2, ks, nt, 4), c);
        store_tile_bf16(sH1 + nt * 16, 64, c, false);  // staged new node tile
      }
    }
    __syncthreads();
    // writeback new node_h (dword copies)
    for (int i = tid; i < NPG * 32; i += 256) {  // dword index
      const int r = i >> 5, cd = i & 31;
      ((unsigned*)sNode)[i] =
          ((const unsigned*)(sHall + (r >> 4) * (2 * 16 * 128) + 16 * 128))
              [(r & 15) * 32 + cd];
    }
    __syncthreads();
  }

  // -------- final edge embeddings: mlpF(fwd) + mlpR(rev) -> global bf16 --------
  for (int t = wave; t < 16; t += 8) {
    build_edge_A(sA, sNode, sEdge, sFrom, sTo, t);
    mlp_layer1_192(sH0, sA, P.Wm1, P.bm1);
    build_edge_A(sA, sNode, sEdge, sTo, sFrom, t);
    mlp_layer1_192(sH1, sA, P.Wr1, P.br1);

    v16bf h0[4], h1[4];
#pragma unroll
    for (int ks = 0; ks < 4; ++ks) {
      h0[ks] = a_frag(sH0 + ks * 32, 128);
      h1[ks] = a_frag(sH1 + ks * 32, 128);
    }
#pragma unroll
    for (int nt = 0; nt < 8; ++nt) {
      v8f c = bias_frag(P.bm2 + nt * 16);
      {
        v8f c2 = bias_frag(P.br2 + nt * 16);
#pragma unroll
        for (int r = 0; r < 8; ++r) c[r] += c2[r];
      }
#pragma unroll
      for (int ks = 0; ks < 4; ++ks) {
        c = wmma_bf16(h0[ks], b_frag_p(P.Wm2, ks, nt, 8), c);
        c = wmma_bf16(h1[ks], b_frag_p(P.Wr2, ks, nt, 8), c);
      }
      __bf16* dst = P.edge_emb + (size_t)(g * EPG + t * 16) * 128 + nt * 16;
      store_tile_bf16(dst, 128, c, false);
    }
  }
}

// ---------------------------------------------------------------------------
// Kernel B: one workgroup per graph pair — fc MLP, S = tq tc^T / TEMP,
// Sinkhorn via u/v potentials, plan materialized in place, WMMA transport
// ---------------------------------------------------------------------------
struct MatchParams {
  const __bf16* edge_emb;
  const __bf16 *Wfc1, *Wfc2;  // frag-packed
  const float *bfc1, *bfc2;
  float* out;
};

constexpr int B_OFF_S = 0;            // bf16[304*320]
constexpr int B_OFF_T = 194560;       // bf16[320*128] (tq|tc, later packed c_emb)
constexpr int B_OFF_U = 276480;       // float[320]
constexpr int B_OFF_V = 277760;       // float[320]
constexpr int B_OFF_R = 279040;       // float[1]
constexpr int B_SMEM  = 279072;

__global__ void __launch_bounds__(256)
match_kernel(MatchParams P) {
  extern __shared__ char smem[];
  __bf16* S = (__bf16*)(smem + B_OFF_S);
  __bf16* T = (__bf16*)(smem + B_OFF_T);
  float* u = (float*)(smem + B_OFF_U);
  float* v = (float*)(smem + B_OFF_V);
  float* red = (float*)(smem + B_OFF_R);

  const int p = blockIdx.x;
  const int tid = threadIdx.x;
  const int wave = tid >> 5;

  // -------- phase 1: tq / tc = fc2(relu(fc1(emb))) --------
  for (int s = 0; s < 2; ++s) {
    const __bf16* emb = P.edge_emb + (size_t)(2 * p + s) * EPG * 128;
    __bf16* Tdst = T + s * (MAXS_T * 64);
    __bf16* Htmp = S + wave * (16 * 64);  // scratch inside S region
    for (int t = wave; t < 16; t += 8) {
      v16bf ea[4];
#pragma unroll
      for (int ks = 0; ks < 4; ++ks)
        ea[ks] = a_frag(emb + t * 16 * 128 + ks * 32, 128);
#pragma unroll
      for (int nt = 0; nt < 4; ++nt) {
        v8f c = bias_frag(P.bfc1 + nt * 16);
#pragma unroll
        for (int ks = 0; ks < 4; ++ks)
          c = wmma_bf16(ea[ks], b_frag_p(P.Wfc1, ks, nt, 4), c);
        store_tile_bf16(Htmp + nt * 16, 64, c, true);
      }
      v16bf ha[2];
#pragma unroll
      for (int ks = 0; ks < 2; ++ks) ha[ks] = a_frag(Htmp + ks * 32, 64);
#pragma unroll
      for (int nt = 0; nt < 4; ++nt) {
        v8f c = bias_frag(P.bfc2 + nt * 16);
#pragma unroll
        for (int ks = 0; ks < 2; ++ks)
          c = wmma_bf16(ha[ks], b_frag_p(P.Wfc2, ks, nt, 4), c);
        store_tile_bf16(Tdst + t * 16 * 64 + nt * 16, 64, c, false);
      }
    }
  }
  // zero padded rows 256..303 of tq/tc (counts==EPG, MAX_SET=300)
  for (int i = tid; i < 2 * 48 * 64; i += 256) {
    const int s = i / (48 * 64), rr = i % (48 * 64);
    T[s * (MAXS_T * 64) + 256 * 64 + rr] = (__bf16)0.0f;
  }
  __syncthreads();

  // -------- phase 2: S = tq @ tc^T / TEMP (pads = -1e30) --------
  const __bf16 NEG = (__bf16)(-1.0e30f);
  for (int i = tid; i < MAXS_T * S_LD; i += 256) S[i] = NEG;
  __syncthreads();
  const __bf16* tq = T;
  const __bf16* tc = T + MAXS_T * 64;
  for (int idx = wave; idx < 19 * 19; idx += 8) {
    const int mt = idx / 19, nt = idx % 19;
    v8f c = {};
#pragma unroll
    for (int ks = 0; ks < 2; ++ks)
      c = wmma_bf16(a_frag(tq + mt * 16 * 64 + ks * 32, 64),
                    bT_frag(tc + nt * 16 * 64 + ks * 32, 64), c);
    const int m0 = (lane_id() >> 4) << 3, n = lane_id() & 15;
#pragma unroll
    for (int r = 0; r < 8; ++r)
      S[(mt * 16 + m0 + r) * S_LD + nt * 16 + n] = (__bf16)(c[r] * 10.0f);  // /TEMP
  }
  __syncthreads();
  for (int i = tid; i < 4 * S_LD; i += 256)
    S[(300 + i / S_LD) * S_LD + (i % S_LD)] = NEG;          // pad rows 300..303
  for (int i = tid; i < 300 * 4; i += 256)
    S[(i / 4) * S_LD + 300 + (i & 3)] = NEG;                // pad cols 300..303
  for (int i = tid; i < S_LD; i += 256) { u[i] = 0.0f; v[i] = 0.0f; }
  __syncthreads();

  // -------- phase 3: Sinkhorn (log_alpha = S - u[q] - v[c]) --------
  for (int it = 0; it < 10; ++it) {
    for (int q = tid; q < 300; q += 256) {
      const __bf16* row = S + q * S_LD;
      float mx = -3.0e38f;
      for (int c = 0; c < 300; ++c) mx = fmaxf(mx, (float)row[c] - v[c]);
      float sm = 0.0f;
      for (int c = 0; c < 300; ++c) sm += __expf((float)row[c] - v[c] - mx);
      u[q] = mx + __logf(sm);
    }
    __syncthreads();
    for (int c = tid; c < 300; c += 256) {
      float mx = -3.0e38f;
      for (int q = 0; q < 300; ++q) mx = fmaxf(mx, (float)S[q * S_LD + c] - u[q]);
      float sm = 0.0f;
      for (int q = 0; q < 300; ++q) sm += __expf((float)S[q * S_LD + c] - u[q] - mx);
      v[c] = mx + __logf(sm);
    }
    __syncthreads();
  }

  // -------- phase 4a: stage c_emb into B-fragment-packed LDS (pads zero) ----
  const __bf16* cembg = P.edge_emb + (size_t)(2 * p + 1) * EPG * 128;
  for (int idx = tid; idx < 10 * 8 * 32 * 16; idx += 256) {
    const int i = idx & 15;
    const int ln = (idx >> 4) & 31;
    const int tile = idx >> 9;  // kt*8 + nt
    const int kt = tile >> 3, nt = tile & 7;
    const int k = kt * 32 + ((ln >> 4) << 4) + i;  // c index 0..319
    const int n = nt * 16 + (ln & 15);             // d index
    T[idx] = (k < 256) ? cembg[k * 128 + n] : (__bf16)0.0f;
  }
  // -------- phase 4b: plan = exp(S - u - v), in place over S (pads -> 0) ----
  for (int i = tid; i < MAXS_T * S_LD; i += 256) {
    const int q = i / S_LD, c = i % S_LD;
    S[i] = (__bf16)__expf((float)S[i] - u[q] - v[c]);
  }
  __syncthreads();

  // -------- phase 5: scores = -sum relu(q_emb - plan @ c_emb) --------
  const __bf16* qembg = P.edge_emb + (size_t)(2 * p) * EPG * 128;
  float lsum = 0.0f;
  for (int idx = wave; idx < 19 * 8; idx += 8) {
    const int mt = idx >> 3, nt = idx & 7;
    v8f acc = {};
#pragma unroll
    for (int ks = 0; ks < 10; ++ks)
      acc = wmma_bf16(a_frag(S + mt * 16 * S_LD + ks * 32, S_LD),  // plan frag
                      frag_lds_p(T, ks * 8 + nt), acc);            // c_emb frag
    const int m0 = (lane_id() >> 4) << 3, n = lane_id() & 15;
    const int dcol = nt * 16 + n;
#pragma unroll
    for (int r = 0; r < 8; ++r) {
      const int q = mt * 16 + m0 + r;
      if (q < 300) {
        const float qe = (q < 256) ? (float)qembg[q * 128 + dcol] : 0.0f;
        lsum += fmaxf(qe - acc[r], 0.0f);
      }
    }
  }
  if (tid == 0) *red = 0.0f;
  __syncthreads();
  atomicAdd(red, lsum);
  __syncthreads();
  if (tid == 0) P.out[p] = -(*red);
}

// ---------------------------------------------------------------------------
// Kernel C: pack f32 weights (K x N row-major) into B-fragment-packed bf16:
// dst[((kt*(N/16)+nt)*32 + lane)*16 + i] = W[kt*32 + (lane>=16)*16 + i][nt*16 + lane%16]
// ---------------------------------------------------------------------------
__global__ void pack_b_kernel(const float* src, __bf16* dst, int Ksrc, int Kpad,
                              int N) {
  const int idx = blockIdx.x * blockDim.x + threadIdx.x;
  if (idx >= Kpad * N) return;
  const int i = idx & 15;
  const int ln = (idx >> 4) & 31;
  const int tile = idx >> 9;
  const int ntiles = N >> 4;
  const int kt = tile / ntiles, nt = tile % ntiles;
  const int k = kt * 32 + ((ln >> 4) << 4) + i;
  const int n = nt * 16 + (ln & 15);
  dst[idx] = (k < Ksrc) ? (__bf16)src[k * N + n] : (__bf16)0.0f;
}

// Workspace layout (bf16 element offsets for packed weights)
constexpr int OFF_WEN  = 0;       // 32x64
constexpr int OFF_WEE  = 2048;    // 32x64 (rows 16..31 zero pad)
constexpr int OFF_WM1  = 4096;    // 192x128
constexpr int OFF_WM2  = 28672;   // 128x128
constexpr int OFF_WR1  = 45056;   // 192x128
constexpr int OFF_WR2  = 69632;   // 128x128
constexpr int OFF_WU1  = 86016;   // 192x128
constexpr int OFF_WU2  = 110592;  // 128x64
constexpr int OFF_WFC1 = 118784;  // 128x64
constexpr int OFF_WFC2 = 126976;  // 64x64
constexpr size_t OFF_EDGE_EMB_BYTES = (size_t)1 << 20;  // edge_emb: E*128 bf16

extern "C" void kernel_launch(void* const* d_in, const int* in_sizes, int n_in,
                              void* d_out, int out_size, void* d_ws, size_t ws_size,
                              hipStream_t stream) {
  (void)in_sizes; (void)n_in; (void)out_size; (void)ws_size;

  __bf16* wsb = (__bf16*)d_ws;
  __bf16* edge_emb = (__bf16*)((char*)d_ws + OFF_EDGE_EMB_BYTES);

  // d_in: 0 node_f, 1 edge_f, 2 from, 3 to, 4 graph_idx(unused),
  // then params (w,b)x: enc_node, enc_edge, msg1, msg2, rmsg1, rmsg2, upd1,
  // upd2, fc1, fc2
  struct Pk { int src; int off; int Ksrc; int Kpad; int N; };
  const Pk pk[10] = {
      {5,  OFF_WEN,  32,  32,  64},
      {7,  OFF_WEE,  16,  32,  64},   // K padded 16 -> 32
      {9,  OFF_WM1,  192, 192, 128},
      {11, OFF_WM2,  128, 128, 128},
      {13, OFF_WR1,  192, 192, 128},
      {15, OFF_WR2,  128, 128, 128},
      {17, OFF_WU1,  192, 192, 128},
      {19, OFF_WU2,  128, 128, 64},
      {21, OFF_WFC1, 128, 128, 64},
      {23, OFF_WFC2, 64,  64,  64},
  };
  for (int i = 0; i < 10; ++i) {
    const int ntot = pk[i].Kpad * pk[i].N;
    pack_b_kernel<<<(ntot + 255) / 256, 256, 0, stream>>>(
        (const float*)d_in[pk[i].src], wsb + pk[i].off, pk[i].Ksrc, pk[i].Kpad,
        pk[i].N);
  }

  (void)hipFuncSetAttribute((const void*)gnn_graph_kernel,
                            hipFuncAttributeMaxDynamicSharedMemorySize, A_SMEM);
  (void)hipFuncSetAttribute((const void*)match_kernel,
                            hipFuncAttributeMaxDynamicSharedMemorySize, B_SMEM);

  GnnParams P;
  P.node_f = (const float*)d_in[0];
  P.edge_f = (const float*)d_in[1];
  P.from_idx = (const int*)d_in[2];
  P.to_idx = (const int*)d_in[3];
  P.Wen = wsb + OFF_WEN;  P.Wee = wsb + OFF_WEE;
  P.Wm1 = wsb + OFF_WM1;  P.Wm2 = wsb + OFF_WM2;
  P.Wr1 = wsb + OFF_WR1;  P.Wr2 = wsb + OFF_WR2;
  P.Wu1 = wsb + OFF_WU1;  P.Wu2 = wsb + OFF_WU2;
  P.ben = (const float*)d_in[6];   P.bee = (const float*)d_in[8];
  P.bm1 = (const float*)d_in[10];  P.bm2 = (const float*)d_in[12];
  P.br1 = (const float*)d_in[14];  P.br2 = (const float*)d_in[16];
  P.bu1 = (const float*)d_in[18];  P.bu2 = (const float*)d_in[20];
  P.edge_emb = edge_emb;
  gnn_graph_kernel<<<G_GRAPHS, 256, A_SMEM, stream>>>(P);

  MatchParams M;
  M.edge_emb = edge_emb;
  M.Wfc1 = wsb + OFF_WFC1;  M.Wfc2 = wsb + OFF_WFC2;
  M.bfc1 = (const float*)d_in[22];  M.bfc2 = (const float*)d_in[24];
  M.out = (float*)d_out;
  match_kernel<<<G_GRAPHS / 2, 256, B_SMEM, stream>>>(M);
}
